// LlamaAttention_61495341744411
// MI455X (gfx1250) — compile-verified, matrix operand order checked by reading, not run on hardware
//
#include <hip/hip_runtime.h>
#include <stdint.h>

#define T_TOTAL 5152
#define DOFF    5120
#define NDEC    32
#define PAST    2048
#define HID     4096
#define NQ      32
#define NKV     8
#define HD      128
#define KVW     (NKV * HD)      // 1024
#define SCALE   0.08838834764831845f
#define LN10000_OVER_HALFD (9.210340371976184f / 128.0f)

typedef __bf16 bf16_t;
typedef bf16_t v16bf __attribute__((ext_vector_type(16)));
typedef float  v8f   __attribute__((ext_vector_type(8)));
typedef unsigned int uv4 __attribute__((ext_vector_type(4)));
typedef int          iv4 __attribute__((ext_vector_type(4)));
typedef int          iv8 __attribute__((ext_vector_type(8)));

union FragU { uint4 u[2]; v16bf v; };

__device__ __forceinline__ unsigned short f2bf(float f) {
  unsigned int u = __float_as_uint(f);
  u += 0x7FFFu + ((u >> 16) & 1u);           // round-to-nearest-even
  return (unsigned short)(u >> 16);
}

// Load a 16x32 bf16 A/B fragment slice for this lane (CDNA5 16-bit layout):
// elements 0..7 at +0, elements 8..15 at +16 halves; caller adds the
// half-wave K-base (0 or 8 halves).
__device__ __forceinline__ v16bf load_frag(const unsigned short* p) {
  FragU f;
  f.u[0] = *(const uint4*)(p);
  f.u[1] = *(const uint4*)(p + 16);
  return f.v;
}

__device__ __forceinline__ unsigned lds_off(const void* p) {
  return (unsigned)(unsigned long long)p;    // LDS aperture keeps offset in low 32 bits
}

// ---------------------------------------------------------------------------
// Tensor Data Mover: 2-D tile load, bf16 elements (data_size=2B).
// D# group0/group1 packed per CDNA5 ISA.
//   line   = tile_d0 contiguous elements, tile_d1 lines
//   lines separated by stride_d0 elements in global memory
// Rows/cols beyond tensor_d0/tensor_d1 read as zero (OOB -> 0).
// 6-arg builtin on this toolchain:
//   (uint32x4 g0, int32x8 g1, int32x4 g2, int32x4 g3, int32x8 pad, i32 cpol)
// ---------------------------------------------------------------------------
__device__ __forceinline__ void tdm_load_2d(unsigned lds_byte_addr,
                                            const void* gptr,
                                            unsigned tile_d0, unsigned tile_d1,
                                            unsigned tensor_d0, unsigned tensor_d1,
                                            unsigned long long stride_d0) {
  unsigned long long ga = (unsigned long long)gptr;
  uv4 g0;
  g0[0] = 1u;                                           // count=1, user-mode
  g0[1] = lds_byte_addr;                                // lds_addr [63:32]
  g0[2] = (unsigned)(ga & 0xffffffffull);               // global_addr[31:0]
  g0[3] = (unsigned)((ga >> 32) & 0x1ffffffull)         // global_addr[56:32]
          | 0x80000000u;                                // type=2 ("image")
  iv8 g1;
  g1[0] = (int)(1u << 16);                              // data_size=1 (2 bytes), wg_mask=0
  g1[1] = (int)((tensor_d0 & 0xffffu) << 16);           // tensor_dim0[15:0] @ [63:48]
  g1[2] = (int)(((tensor_d0 >> 16) & 0xffffu) |
                ((tensor_d1 & 0xffffu) << 16));         // dim0 hi | dim1 lo
  g1[3] = (int)(((tensor_d1 >> 16) & 0xffffu) |
                ((tile_d0 & 0xffffu) << 16));           // dim1 hi | tile_dim0
  g1[4] = (int)(tile_d1 & 0xffffu);                     // tile_dim1 (tile_dim2=0)
  g1[5] = (int)(stride_d0 & 0xffffffffull);             // tensor_dim0_stride lo
  g1[6] = (int)((stride_d0 >> 32) & 0xffffull);         // stride hi (dim1_stride=0)
  g1[7] = 0;
  iv4 z4 = {0, 0, 0, 0};
  iv8 z8 = {0, 0, 0, 0, 0, 0, 0, 0};
  __builtin_amdgcn_tensor_load_to_lds(g0, g1, z4, z4, z8, 0);
}

// ---------------------------------------------------------------------------
// GEMM: C[M,N] = A[M,K] * B[N,K]^T, bf16 inputs, fp32 WMMA accumulate.
// Block 128x128, 8 waves, wave tile 32x64 (2x4 frags), BK=64.
// TDM double-buffered staging: wave0 issues stage k+1 while stage k computes.
// All 6 fragments of a K-chunk are loaded into distinct registers before the
// WMMA chain so DS latency overlaps with matrix issue (partial dscnt waits).
// ---------------------------------------------------------------------------
__global__ __launch_bounds__(256)
void gemm_bf16_tdm(const unsigned short* __restrict__ A,
                   const unsigned short* __restrict__ B,
                   float* __restrict__ C, int M, int N, int K) {
  __shared__ __align__(16) unsigned short As[2][128 * 64];
  __shared__ __align__(16) unsigned short Bs[2][128 * 64];
  const int tid  = threadIdx.x;
  const int lane = tid & 31;
  const int wave = tid >> 5;
  const int wv   = __builtin_amdgcn_readfirstlane(tid) >> 5;  // scalar wave id
  const int m0 = blockIdx.y * 128;
  const int n0 = blockIdx.x * 128;
  const int wm = (wave >> 1) * 32;   // 0,32,64,96
  const int wn = (wave & 1) * 64;    // 0,64
  const int kb = (lane >> 4) * 8;
  const int ln = lane & 15;
  const int nstages = K >> 6;
  const unsigned rowsA = (unsigned)(M - m0);

  v8f acc[2][4] = {};

  if (wv == 0) {
    tdm_load_2d(lds_off(&As[0][0]), A + (size_t)m0 * K, 64, 128, (unsigned)K, rowsA, (unsigned)K);
    tdm_load_2d(lds_off(&Bs[0][0]), B + (size_t)n0 * K, 64, 128, (unsigned)K, 128u, (unsigned)K);
  }

  for (int st = 0; st < nstages; ++st) {
    const int buf = st & 1;
    if (wv == 0) {
      if (st + 1 < nstages) {
        const int k1 = (st + 1) << 6;
        tdm_load_2d(lds_off(&As[buf ^ 1][0]), A + (size_t)m0 * K + k1, 64, 128,
                    (unsigned)(K - k1), rowsA, (unsigned)K);
        tdm_load_2d(lds_off(&Bs[buf ^ 1][0]), B + (size_t)n0 * K + k1, 64, 128,
                    (unsigned)(K - k1), 128u, (unsigned)K);
        __builtin_amdgcn_s_wait_tensorcnt(2);   // retire only current stage's pair
      } else {
        __builtin_amdgcn_s_wait_tensorcnt(0);
      }
    }
    __syncthreads();
    const unsigned short* Ab = &As[buf][0];
    const unsigned short* Bb = &Bs[buf][0];
    #pragma unroll
    for (int c = 0; c < 2; ++c) {
      // batch all fragment loads first: distinct live registers -> staggered waits
      v16bf a0 = load_frag(Ab + (wm +      ln) * 64 + c * 32 + kb);
      v16bf a1 = load_frag(Ab + (wm + 16 + ln) * 64 + c * 32 + kb);
      v16bf b0 = load_frag(Bb + (wn +  0 + ln) * 64 + c * 32 + kb);
      v16bf b1 = load_frag(Bb + (wn + 16 + ln) * 64 + c * 32 + kb);
      v16bf b2 = load_frag(Bb + (wn + 32 + ln) * 64 + c * 32 + kb);
      v16bf b3 = load_frag(Bb + (wn + 48 + ln) * 64 + c * 32 + kb);
      acc[0][0] = __builtin_amdgcn_wmma_f32_16x16x32_bf16(false, a0, false, b0, (short)0, acc[0][0], false, false);
      acc[1][0] = __builtin_amdgcn_wmma_f32_16x16x32_bf16(false, a1, false, b0, (short)0, acc[1][0], false, false);
      acc[0][1] = __builtin_amdgcn_wmma_f32_16x16x32_bf16(false, a0, false, b1, (short)0, acc[0][1], false, false);
      acc[1][1] = __builtin_amdgcn_wmma_f32_16x16x32_bf16(false, a1, false, b1, (short)0, acc[1][1], false, false);
      acc[0][2] = __builtin_amdgcn_wmma_f32_16x16x32_bf16(false, a0, false, b2, (short)0, acc[0][2], false, false);
      acc[1][2] = __builtin_amdgcn_wmma_f32_16x16x32_bf16(false, a1, false, b2, (short)0, acc[1][2], false, false);
      acc[0][3] = __builtin_amdgcn_wmma_f32_16x16x32_bf16(false, a0, false, b3, (short)0, acc[0][3], false, false);
      acc[1][3] = __builtin_amdgcn_wmma_f32_16x16x32_bf16(false, a1, false, b3, (short)0, acc[1][3], false, false);
    }
    __syncthreads();
  }

  #pragma unroll
  for (int i = 0; i < 2; ++i)
    #pragma unroll
    for (int r = 0; r < 8; ++r) {
      int m = m0 + wm + i * 16 + r + ((lane >> 4) << 3);
      if (m < M) {
        float* cp = C + (size_t)m * N + n0 + wn + ln;
        cp[0]  = acc[i][0][r];
        cp[16] = acc[i][1][r];
        cp[32] = acc[i][2][r];
        cp[48] = acc[i][3][r];
      }
    }
}

// ---------------------------------------------------------------------------
// fp32 -> bf16, 8 elements per thread (all sizes are multiples of 8).
// ---------------------------------------------------------------------------
__global__ void tobf8_kernel(const float* __restrict__ in,
                             unsigned short* __restrict__ out, int total8) {
  int idx = blockIdx.x * blockDim.x + threadIdx.x;
  if (idx >= total8) return;
  const float4* p = (const float4*)in + (size_t)idx * 2;
  float4 f0 = p[0], f1 = p[1];
  uint4 o;
  o.x = (unsigned)f2bf(f0.x) | ((unsigned)f2bf(f0.y) << 16);
  o.y = (unsigned)f2bf(f0.z) | ((unsigned)f2bf(f0.w) << 16);
  o.z = (unsigned)f2bf(f1.x) | ((unsigned)f2bf(f1.y) << 16);
  o.w = (unsigned)f2bf(f1.z) | ((unsigned)f2bf(f1.w) << 16);
  ((uint4*)out)[idx] = o;
}

__device__ __forceinline__ float pos_of_token(int t) {
  if (t < 1024) return (float)t;
  if (t < 2560) return (float)(t - 1024);
  if (t < 4608) return (float)(t - 2560);
  if (t < 5120) return (float)(t - 4608);
  return (float)PAST;
}

// RoPE: fp32 in -> bf16 out (all tokens) + fp32 out for decode tokens.
__global__ void rope_kernel(const float* __restrict__ in,
                            unsigned short* __restrict__ out_bf,
                            float* __restrict__ out_dec,
                            int NH, int total) {
  int idx = blockIdx.x * blockDim.x + threadIdx.x;
  if (idx >= total) return;
  int i = idx & 63;
  int h = (idx >> 6) % NH;
  int t = idx / (64 * NH);
  float pos = pos_of_token(t);
  float inv = __expf(-(float)(2 * i) * LN10000_OVER_HALFD);
  float ang = pos * inv;
  float c = __cosf(ang), s = __sinf(ang);
  size_t base = (size_t)t * NH * HD + (size_t)h * HD;
  float x0 = in[base + i], x1 = in[base + i + 64];
  float o0 = x0 * c - x1 * s;
  float o1 = x1 * c + x0 * s;
  out_bf[base + i]      = f2bf(o0);
  out_bf[base + i + 64] = f2bf(o1);
  if (t >= DOFF) {
    size_t db = (size_t)(t - DOFF) * NH * HD + (size_t)h * HD;
    out_dec[db + i]      = o0;
    out_dec[db + i + 64] = o1;
  }
}

// V pre-transpose: vtb[(kvh*HD + dim) * T_TOTAL + token] = bf16(v[token][kvh][dim])
// so prefill V tiles become contiguous lines for the TDM.
__global__ void v_transpose_bf(const float* __restrict__ vf,
                               unsigned short* __restrict__ vtb) {
  int idx = blockIdx.x * blockDim.x + threadIdx.x;
  if (idx >= KVW * T_TOTAL) return;
  int t = idx % T_TOTAL;
  int c = idx / T_TOTAL;
  vtb[idx] = f2bf(vf[(size_t)t * KVW + c]);
}

// ---------------------------------------------------------------------------
// Prefill flash attention. Grid (NQ, 40). 8 waves x 16 q-rows. K tiles of 32.
// K and V tiles staged by the TDM; S=Q*K^T and O+=P*V via WMMA.
// ---------------------------------------------------------------------------
__global__ __launch_bounds__(256)
void prefill_attn(const unsigned short* __restrict__ qbf,
                  const unsigned short* __restrict__ kbf,
                  const unsigned short* __restrict__ vtb,
                  float* __restrict__ attn) {
  __shared__ __align__(16) unsigned short Ks[32 * 128];    // [key][dim]
  __shared__ __align__(16) unsigned short Vt[128 * 32];    // [dim][key]
  __shared__ __align__(16) unsigned short Ps[8 * 16 * 32]; // per-wave P staging
  const int h   = blockIdx.x;
  const int kvh = h >> 2;
  int tgt = blockIdx.y, seqoff, q0;
  if      (tgt <  8) { seqoff = 0;    q0 = tgt * 128; }
  else if (tgt < 20) { seqoff = 1024; q0 = (tgt -  8) * 128; }
  else if (tgt < 36) { seqoff = 2560; q0 = (tgt - 20) * 128; }
  else               { seqoff = 4608; q0 = (tgt - 36) * 128; }
  const int tid  = threadIdx.x, lane = tid & 31, wave = tid >> 5;
  const int wv   = __builtin_amdgcn_readfirstlane(tid) >> 5;
  const int qb = q0 + wave * 16;
  const int kb = (lane >> 4) * 8;
  const int ln = lane & 15;

  v16bf qf[4];
  {
    const unsigned short* qp = qbf + (size_t)(seqoff + qb + ln) * HID + h * HD + kb;
    #pragma unroll
    for (int c = 0; c < 4; ++c) qf[c] = load_frag(qp + c * 32);
  }

  v8f O[8] = {};
  float rmax[8], rsum[8];
  #pragma unroll
  for (int r = 0; r < 8; ++r) { rmax[r] = -3.0e38f; rsum[r] = 0.f; }
  unsigned short* Pw = Ps + wave * (16 * 32);

  for (int k0 = 0; k0 < q0 + 128; k0 += 32) {
    __syncthreads();                       // previous tile fully consumed
    if (wv == 0) {
      tdm_load_2d(lds_off(&Ks[0]),
                  kbf + (size_t)(seqoff + k0) * KVW + kvh * HD,
                  128, 32, 128, 32, (unsigned long long)KVW);
      tdm_load_2d(lds_off(&Vt[0]),
                  vtb + (size_t)(kvh * HD) * T_TOTAL + seqoff + k0,
                  32, 128, 32, 128, (unsigned long long)T_TOTAL);
      __builtin_amdgcn_s_wait_tensorcnt(0);
    }
    __syncthreads();

    // S[16 x 32] = Q[16 x 128] * K^T ; preload all 8 B-frags, then two
    // interleaved 4-deep accumulator chains.
    v8f s0 = {}, s1 = {};
    {
      v16bf kb0[4], kb1[4];
      #pragma unroll
      for (int c = 0; c < 4; ++c) {
        kb0[c] = load_frag(Ks + ln * 128 + c * 32 + kb);
        kb1[c] = load_frag(Ks + (ln + 16) * 128 + c * 32 + kb);
      }
      #pragma unroll
      for (int c = 0; c < 4; ++c) {
        s0 = __builtin_amdgcn_wmma_f32_16x16x32_bf16(false, qf[c], false, kb0[c], (short)0, s0, false, false);
        s1 = __builtin_amdgcn_wmma_f32_16x16x32_bf16(false, qf[c], false, kb1[c], (short)0, s1, false, false);
      }
    }

    // online softmax: row r lives in one half-wave (16 lanes) at C-vgpr r
    #pragma unroll
    for (int r = 0; r < 8; ++r) {
      int m = qb + r + ((lane >> 4) << 3);
      float x0 = s0[r] * SCALE;
      float x1 = s1[r] * SCALE;
      if (k0 + ln      > m) x0 = -1.0e30f;
      if (k0 + ln + 16 > m) x1 = -1.0e30f;
      float v = fmaxf(x0, x1);
      #pragma unroll
      for (int off = 1; off < 16; off <<= 1) v = fmaxf(v, __shfl_xor(v, off, 32));
      float nm = fmaxf(rmax[r], v);
      float so = __expf(rmax[r] - nm);
      rmax[r] = nm;
      float p0 = __expf(x0 - nm), p1 = __expf(x1 - nm);
      float rs = p0 + p1;
      #pragma unroll
      for (int off = 1; off < 16; off <<= 1) rs += __shfl_xor(rs, off, 32);
      rsum[r] = rsum[r] * so + rs;
      #pragma unroll
      for (int j = 0; j < 8; ++j) O[j][r] *= so;
      int mrow = r + ((lane >> 4) << 3);
      Pw[mrow * 32 + ln]      = f2bf(p0);
      Pw[mrow * 32 + ln + 16] = f2bf(p1);
    }

    // wave-internal LDS RAW: P written by all lanes, re-read in A-frag layout
    __builtin_amdgcn_wave_barrier();
    asm volatile("s_wait_dscnt 0" ::: "memory");
    v16bf pf = load_frag(Pw + ln * 32 + kb);

    // O[16 x 128] += P[16 x 32] * V[32 x 128] (independent chains per j)
    #pragma unroll
    for (int j = 0; j < 8; ++j) {
      v16bf bv = load_frag(Vt + (j * 16 + ln) * 32 + kb);
      O[j] = __builtin_amdgcn_wmma_f32_16x16x32_bf16(false, pf, false, bv, (short)0, O[j], false, false);
    }
  }

  #pragma unroll
  for (int j = 0; j < 8; ++j)
    #pragma unroll
    for (int r = 0; r < 8; ++r) {
      int m = qb + r + ((lane >> 4) << 3);
      attn[(size_t)(seqoff + m) * HID + h * HD + j * 16 + ln] = O[j][r] / rsum[r];
    }
}

// ---------------------------------------------------------------------------
// Decode attention: 32 tokens x 8 kv-heads over 2049 keys, cached keys get
// RoPE on the fly; two-pass softmax via LDS. VALU-bound but tiny.
// ---------------------------------------------------------------------------
__global__ __launch_bounds__(256)
void decode_attn(const float* __restrict__ qdec, const float* __restrict__ kdec,
                 const float* __restrict__ vnew, const float* __restrict__ kcache,
                 const float* __restrict__ vcache, float* __restrict__ attn) {
  __shared__ float qs[4 * HD];
  __shared__ float sc[4 * 2052];
  __shared__ float red[4 * 256];
  __shared__ float invf[64];
  const int d   = blockIdx.x >> 3;
  const int kh  = blockIdx.x & 7;
  const int tid = threadIdx.x;
  if (tid < 64) invf[tid] = __expf(-(float)(2 * tid) * LN10000_OVER_HALFD);
  for (int v = tid; v < 4 * HD; v += 256)
    qs[v] = qdec[(size_t)d * (NQ * HD) + (size_t)(kh * 4) * HD + v];
  __syncthreads();

  const int NPOS = PAST + 1;
  for (int p = tid; p < NPOS; p += 256) {
    float d0 = 0, d1 = 0, d2 = 0, d3 = 0;
    if (p < PAST) {
      const float* kp = kcache + (size_t)d * PAST * KVW + (size_t)p * KVW + kh * HD;
      float fp = (float)p;
      for (int i = 0; i < 64; ++i) {
        float ang = fp * invf[i];
        float c = __cosf(ang), s = __sinf(ang);
        float x0 = kp[i], x1 = kp[i + 64];
        float r0 = x0 * c - x1 * s;
        float r1 = x1 * c + x0 * s;
        d0 += r0 * qs[0 * HD + i] + r1 * qs[0 * HD + i + 64];
        d1 += r0 * qs[1 * HD + i] + r1 * qs[1 * HD + i + 64];
        d2 += r0 * qs[2 * HD + i] + r1 * qs[2 * HD + i + 64];
        d3 += r0 * qs[3 * HD + i] + r1 * qs[3 * HD + i + 64];
      }
    } else {
      const float* kp = kdec + (size_t)d * KVW + kh * HD;   // RoPE'd @ PAST already
      for (int i = 0; i < HD; ++i) {
        float kx = kp[i];
        d0 += kx * qs[0 * HD + i];
        d1 += kx * qs[1 * HD + i];
        d2 += kx * qs[2 * HD + i];
        d3 += kx * qs[3 * HD + i];
      }
    }
    sc[0 * 2052 + p] = d0 * SCALE;
    sc[1 * 2052 + p] = d1 * SCALE;
    sc[2 * 2052 + p] = d2 * SCALE;
    sc[3 * 2052 + p] = d3 * SCALE;
  }
  __syncthreads();

  float pm[4] = {-3.0e38f, -3.0e38f, -3.0e38f, -3.0e38f};
  for (int p = tid; p < NPOS; p += 256)
    #pragma unroll
    for (int g = 0; g < 4; ++g) pm[g] = fmaxf(pm[g], sc[g * 2052 + p]);
  #pragma unroll
  for (int g = 0; g < 4; ++g) red[g * 256 + tid] = pm[g];
  __syncthreads();
  for (int s = 128; s > 0; s >>= 1) {
    if (tid < s)
      #pragma unroll
      for (int g = 0; g < 4; ++g)
        red[g * 256 + tid] = fmaxf(red[g * 256 + tid], red[g * 256 + tid + s]);
    __syncthreads();
  }
  float mx[4];
  #pragma unroll
  for (int g = 0; g < 4; ++g) mx[g] = red[g * 256];
  __syncthreads();

  float ps[4] = {0.f, 0.f, 0.f, 0.f};
  for (int p = tid; p < NPOS; p += 256)
    #pragma unroll
    for (int g = 0; g < 4; ++g) {
      float e = __expf(sc[g * 2052 + p] - mx[g]);
      sc[g * 2052 + p] = e;
      ps[g] += e;
    }
  __syncthreads();
  #pragma unroll
  for (int g = 0; g < 4; ++g) red[g * 256 + tid] = ps[g];
  __syncthreads();
  for (int s = 128; s > 0; s >>= 1) {
    if (tid < s)
      #pragma unroll
      for (int g = 0; g < 4; ++g)
        red[g * 256 + tid] += red[g * 256 + tid + s];
    __syncthreads();
  }
  float sm[4];
  #pragma unroll
  for (int g = 0; g < 4; ++g) sm[g] = red[g * 256];

  #pragma unroll
  for (int pair = 0; pair < 2; ++pair) {
    int lin = tid + pair * 256;
    int g = lin >> 7;
    int j = lin & 127;
    float o = 0.f;
    const float* vp = vcache + (size_t)d * PAST * KVW + kh * HD + j;
    for (int p = 0; p < PAST; ++p) o += sc[g * 2052 + p] * vp[(size_t)p * KVW];
    o += sc[g * 2052 + PAST] * vnew[(size_t)(DOFF + d) * KVW + kh * HD + j];
    attn[(size_t)(DOFF + d) * HID + (size_t)(kh * 4 + g) * HD + j] = o / sm[g];
  }
}

// ---------------------------------------------------------------------------
extern "C" void kernel_launch(void* const* d_in, const int* in_sizes, int n_in,
                              void* d_out, int out_size, void* d_ws, size_t ws_size,
                              hipStream_t stream) {
  const float* hs = (const float*)d_in[0];
  const float* wq = (const float*)d_in[1];
  const float* wk = (const float*)d_in[2];
  const float* wv = (const float*)d_in[3];
  const float* wo = (const float*)d_in[4];
  const float* kc = (const float*)d_in[5];
  const float* vc = (const float*)d_in[6];
  float* out = (float*)d_out;

  char* ws = (char*)d_ws;
  size_t off = 0;
  auto alloc = [&](size_t bytes) -> void* {
    off = (off + 255) & ~(size_t)255;
    void* p = ws + off;
    off += bytes;
    return p;
  };
  unsigned short* hbf  = (unsigned short*)alloc((size_t)T_TOTAL * HID * 2);
  unsigned short* wqb  = (unsigned short*)alloc((size_t)HID * HID * 2);
  unsigned short* wkb  = (unsigned short*)alloc((size_t)KVW * HID * 2);
  unsigned short* wvb  = (unsigned short*)alloc((size_t)KVW * HID * 2);
  unsigned short* wob  = (unsigned short*)alloc((size_t)HID * HID * 2);
  float*          qf   = (float*)         alloc((size_t)T_TOTAL * HID * 4);
  float*          kf   = (float*)         alloc((size_t)T_TOTAL * KVW * 4);
  float*          vf   = (float*)         alloc((size_t)T_TOTAL * KVW * 4);
  unsigned short* qbf  = (unsigned short*)alloc((size_t)T_TOTAL * HID * 2);
  unsigned short* kbf  = (unsigned short*)alloc((size_t)T_TOTAL * KVW * 2);
  unsigned short* vtb  = (unsigned short*)alloc((size_t)T_TOTAL * KVW * 2);
  float*          attn = (float*)         alloc((size_t)T_TOTAL * HID * 4);
  unsigned short* abf  = (unsigned short*)alloc((size_t)T_TOTAL * HID * 2);
  float*          qdec = (float*)         alloc((size_t)NDEC * HID * 4);
  float*          kdec = (float*)         alloc((size_t)NDEC * KVW * 4);

  dim3 blk(256);
  auto cvt = [&](const float* src, unsigned short* dst, size_t n) {
    int n8 = (int)(n / 8);
    tobf8_kernel<<<(n8 + 255) / 256, blk, 0, stream>>>(src, dst, n8);
  };

  // one-time bf16 conversions (keeps GEMM/TDM traffic at 2 bytes/element)
  cvt(hs, hbf, (size_t)T_TOTAL * HID);
  cvt(wq, wqb, (size_t)HID * HID);
  cvt(wk, wkb, (size_t)KVW * HID);
  cvt(wv, wvb, (size_t)KVW * HID);
  cvt(wo, wob, (size_t)HID * HID);

  const int MT = (T_TOTAL + 127) / 128;
  gemm_bf16_tdm<<<dim3(HID / 128, MT), blk, 0, stream>>>(hbf, wqb, qf, T_TOTAL, HID, HID);
  gemm_bf16_tdm<<<dim3(KVW / 128, MT), blk, 0, stream>>>(hbf, wkb, kf, T_TOTAL, KVW, HID);
  gemm_bf16_tdm<<<dim3(KVW / 128, MT), blk, 0, stream>>>(hbf, wvb, vf, T_TOTAL, KVW, HID);

  {
    int tq = T_TOTAL * NQ * 64;
    rope_kernel<<<(tq + 255) / 256, blk, 0, stream>>>(qf, qbf, qdec, NQ, tq);
    int tk = T_TOTAL * NKV * 64;
    rope_kernel<<<(tk + 255) / 256, blk, 0, stream>>>(kf, kbf, kdec, NKV, tk);
    int tv = T_TOTAL * KVW;
    v_transpose_bf<<<(tv + 255) / 256, blk, 0, stream>>>(vf, vtb);
  }

  prefill_attn<<<dim3(NQ, 40), blk, 0, stream>>>(qbf, kbf, vtb, attn);
  decode_attn<<<dim3(NDEC * NKV), blk, 0, stream>>>(qdec, kdec, vf, kc, vc, attn);

  cvt(attn, abf, (size_t)T_TOTAL * HID);
  gemm_bf16_tdm<<<dim3(HID / 128, MT), blk, 0, stream>>>(abf, wob, out, T_TOTAL, HID, HID);
}
